// model_mamba_autoencoder_31894427140384
// MI455X (gfx1250) — matrix operand order repaired and checked by reading, not compiled, not run on hardware
//
#include <hip/hip_runtime.h>
#include <hip/hip_bf16.h>
#include <math.h>

// ---------------- model dims ----------------
#define TOK     (8 * 2048)   // BATCH * SEQ tokens
#define BATCH_  8
#define SEQLEN  2048
#define DMODEL  512
#define DINNER  1024
#define NSTATE  16
#define DTRANK  32
#define INSZ    64
#define BOTTLE  10

// GEMM tiling
#define MB      32           // rows per block (2 m-subtiles)
#define KP      512          // K panel staged in LDS
#define SASTR   (KP + 8)     // LDS row stride (shorts): 1040B -> bank-spread

// ---------------- WMMA types ----------------
typedef __attribute__((ext_vector_type(16))) __bf16        v16bf;
typedef __attribute__((ext_vector_type(8)))  float         v8f;
typedef __attribute__((ext_vector_type(8)))  unsigned int  v8u;

__device__ __forceinline__ unsigned short f32_to_bf16_rne(float f) {
  union { float f; unsigned int u; } v; v.f = f;
  unsigned int r = v.u + 0x7FFFu + ((v.u >> 16) & 1u);
  return (unsigned short)(r >> 16);
}

__device__ __forceinline__ v16bf pack_frag(uint4 lo, uint4 hi8) {
  v8u u;
  u[0] = lo.x; u[1] = lo.y; u[2] = lo.z; u[3] = lo.w;
  u[4] = hi8.x; u[5] = hi8.y; u[6] = hi8.z; u[7] = hi8.w;
  return __builtin_bit_cast(v16bf, u);
}

// ---------------------------------------------------------------------------
// fp32 -> bf16 conversion with row/col zero padding.
// ---------------------------------------------------------------------------
__global__ __launch_bounds__(256) void cvt_pad_bf16(
    const float* __restrict__ src, unsigned short* __restrict__ dst,
    int rows, int srows, int sstride, int soff, int cols, int dcols)
{
  long long i = (long long)blockIdx.x * blockDim.x + threadIdx.x;
  long long total = (long long)rows * dcols;
  if (i >= total) return;
  int c = (int)(i % dcols);
  long long r = i / dcols;
  float v = 0.0f;
  if (r < srows && c < cols) v = src[r * (long long)sstride + soff + c];
  dst[i] = f32_to_bf16_rne(v);
}

// ---------------------------------------------------------------------------
// C(M,N) fp32 = A(M,K)bf16 @ W(N,K)bf16^T + bias, optional softplus epilogue.
//
// Block = 256 threads = 8 waves; block tile 32 x 512.
// A strip (32 x K) staged in LDS in K-panels of 512 via
// global_load_async_to_lds_b128 (+ s_wait_asynccnt + barrier); fragments read
// back with ds_load_b128. Each wave: 2 m-subtiles x 4 n-subtiles = 8
// accumulators -> each A fragment feeds 4 WMMAs, each B fragment feeds 2.
//
// Per-lane fragment layout (ISA 7.12.2, wave32):
//   A 16x32:  lane L -> row L&15; K halves [h*8,+8) and [h*8+16,+8), h=L>>4
//   B 32x16:  lane L -> col L&15; K halves [h*8,+8) and [h*8+16,+8)
//   C/D:      VGPR v -> row v + h*8, col L&15
// Requires K % 32 == 0, N % 16 == 0, M % 32 == 0 (buffers pre-padded).
// ---------------------------------------------------------------------------
__global__ __launch_bounds__(256) void gemm_bf16_wmma(
    const unsigned short* __restrict__ A,   // (M,K) bf16 row-major
    const unsigned short* __restrict__ W,   // (N,K) bf16 row-major (= B^T)
    const float* __restrict__ bias,         // (biasN) or nullptr
    float* __restrict__ C,                  // (M,N) fp32
    int M, int N, int K, int biasN, int epilogue)
{
  __shared__ unsigned short sA[MB * SASTR];

  const int tid  = threadIdx.x;
  const int wave = tid >> 5;
  const int lane = tid & 31;
  const int l15  = lane & 15;
  const int hi   = (lane >> 4) & 1;
  const int mBase = blockIdx.x * MB;

  // 4 n-subtiles per wave, strided across the 8 waves of the block
  int nt[4];
  bool act[4];
  const unsigned short* wp[4];
  v8f acc[4][2];
#pragma unroll
  for (int j = 0; j < 4; ++j) {
    nt[j]  = blockIdx.y * 32 + j * 8 + wave;
    act[j] = (nt[j] * 16 < N);
    const int col = act[j] ? (nt[j] * 16 + l15) : 0;     // dummy col 0 if idle
    wp[j] = W + (long long)col * K + hi * 8;
    float bv = 0.0f;
    if (bias != nullptr && act[j] && col < biasN) bv = bias[col];
#pragma unroll
    for (int mi = 0; mi < 2; ++mi)
#pragma unroll
      for (int v = 0; v < 8; ++v) acc[j][mi][v] = bv;
  }

  // LDS fragment base pointers for the two m-subtiles
  const unsigned short* sa0 = &sA[l15 * SASTR + hi * 8];
  const unsigned short* sa1 = &sA[(16 + l15) * SASTR + hi * 8];

  for (int kp = 0; kp < K; kp += KP) {
    const int kcur = (K - kp < KP) ? (K - kp) : KP;

    // ---- async copy A panel (MB x kcur) global -> LDS ----
    {
      const int kc8 = kcur >> 3;                 // 16B chunks per row
      const int total = MB * kc8;
      for (int v = tid; v < total; v += 256) {
        const int row = v / kc8;
        const int cv  = v - row * kc8;
        unsigned dst = (unsigned)(size_t)(&sA[row * SASTR + cv * 8]);
        const unsigned short* src =
            A + (long long)(mBase + row) * K + kp + cv * 8;
        asm volatile("global_load_async_to_lds_b128 %0, %1, off"
                     :: "v"(dst), "v"(src) : "memory");
      }
      asm volatile("s_wait_asynccnt 0" ::: "memory");
    }
    __syncthreads();

    // ---- compute over this K panel ----
    for (int kk = 0; kk < kcur; kk += 32) {
      uint4 a00 = *(const uint4*)(sa0 + kk);
      uint4 a01 = *(const uint4*)(sa0 + kk + 16);
      uint4 a10 = *(const uint4*)(sa1 + kk);
      uint4 a11 = *(const uint4*)(sa1 + kk + 16);
      v16bf af0 = pack_frag(a00, a01);
      v16bf af1 = pack_frag(a10, a11);
      const int k = kp + kk;
#pragma unroll
      for (int j = 0; j < 4; ++j) {
        uint4 b0 = *(const uint4*)(wp[j] + k);
        uint4 b1 = *(const uint4*)(wp[j] + k + 16);
        v16bf bf = pack_frag(b0, b1);
        acc[j][0] = __builtin_amdgcn_wmma_f32_16x16x32_bf16(
            false, af0, false, bf, (short)0, acc[j][0], false, false);
        acc[j][1] = __builtin_amdgcn_wmma_f32_16x16x32_bf16(
            false, af1, false, bf, (short)0, acc[j][1], false, false);
      }
    }
    __syncthreads();   // LDS panel reuse
  }

  // ---- epilogue + store ----
#pragma unroll
  for (int j = 0; j < 4; ++j) {
    if (!act[j]) continue;
    const int col = nt[j] * 16 + l15;
#pragma unroll
    for (int mi = 0; mi < 2; ++mi) {
#pragma unroll
      for (int v = 0; v < 8; ++v) {
        int row = mBase + mi * 16 + v + hi * 8;
        float val = acc[j][mi][v];
        if (epilogue == 1) {                   // softplus
          val = (val > 20.0f) ? val : log1pf(__expf(val));
        }
        C[(long long)row * N + col] = val;
      }
    }
  }
}

// ---------------------------------------------------------------------------
// Depthwise causal conv (D_CONV=4) over seq + SiLU.
// ---------------------------------------------------------------------------
__global__ __launch_bounds__(256) void conv_silu_kernel(
    const float* __restrict__ xz, const float* __restrict__ cw,
    const float* __restrict__ cb, float* __restrict__ uc)
{
  long long i = (long long)blockIdx.x * blockDim.x + threadIdx.x;
  if (i >= (long long)TOK * DINNER) return;
  int d = (int)(i & (DINNER - 1));
  long long m = i >> 10;                 // token index = b*SEQ + t
  int t = (int)(m & (SEQLEN - 1));
  float acc = cb[d];
#pragma unroll
  for (int j = 0; j < 4; ++j) {
    int dt = j - 3;                      // taps at t-3 .. t
    if (t + dt >= 0)
      acc += cw[d * 4 + j] * xz[(m + dt) * (2 * DINNER) + d];
  }
  uc[i] = acc / (1.0f + __expf(-acc));   // SiLU
}

// ---------------------------------------------------------------------------
// Selective-scan: sequential over t, parallel over (batch, d_inner).
// ---------------------------------------------------------------------------
__global__ __launch_bounds__(256) void scan_kernel(
    const float* __restrict__ xdbl,    // (TOK,64): cols 32..47 = B, 48..63 = C
    const float* __restrict__ delta,   // (TOK,DINNER) post-softplus
    const float* __restrict__ uc,      // (TOK,DINNER)
    const float* __restrict__ A_log,   // (DINNER,16)
    float* __restrict__ ycore)         // (TOK,DINNER)
{
  __shared__ float sBC[32];
  const int b = blockIdx.x >> 2;                       // DINNER/256 = 4 chunks
  const int d = ((blockIdx.x & 3) << 8) + threadIdx.x;

  float An[NSTATE], h[NSTATE];
#pragma unroll
  for (int n = 0; n < NSTATE; ++n) {
    An[n] = -__expf(A_log[d * NSTATE + n]);
    h[n] = 0.0f;
  }

  const long long mBase = (long long)b * SEQLEN;
  for (int t = 0; t < SEQLEN; ++t) {
    const long long m = mBase + t;
    if (threadIdx.x < 32) sBC[threadIdx.x] = xdbl[m * 64 + 32 + threadIdx.x];
    __syncthreads();
    const float dv = delta[m * DINNER + d];
    const float du = dv * uc[m * DINNER + d];
    float y = 0.0f;
#pragma unroll
    for (int n = 0; n < NSTATE; ++n) {
      h[n] = __expf(dv * An[n]) * h[n] + du * sBC[n];
      y += h[n] * sBC[NSTATE + n];
    }
    ycore[m * DINNER + d] = y;
    __syncthreads();
  }
}

// ---------------------------------------------------------------------------
// y = (ycore + uc*D) * silu(z); emit bf16 for the out-proj GEMM.
// ---------------------------------------------------------------------------
__global__ __launch_bounds__(256) void gate_kernel(
    const float* __restrict__ ycore, const float* __restrict__ uc,
    const float* __restrict__ xz, const float* __restrict__ Dp,
    unsigned short* __restrict__ out)
{
  long long i = (long long)blockIdx.x * blockDim.x + threadIdx.x;
  if (i >= (long long)TOK * DINNER) return;
  int d = (int)(i & (DINNER - 1));
  long long m = i >> 10;
  float z = xz[m * (2 * DINNER) + DINNER + d];
  float y = (ycore[i] + uc[i] * Dp[d]) * (z / (1.0f + __expf(-z)));
  out[i] = f32_to_bf16_rne(y);
}

// ===========================================================================
extern "C" void kernel_launch(void* const* d_in, const int* in_sizes, int n_in,
                              void* d_out, int out_size, void* d_ws, size_t ws_size,
                              hipStream_t stream)
{
  (void)in_sizes; (void)n_in; (void)out_size; (void)ws_size;

  // ---- inputs, setup_inputs() dict order ----
  const float* x          = (const float*)d_in[0];
  const float* fc_in_w    = (const float*)d_in[1];
  const float* fc_in_b    = (const float*)d_in[2];
  const float* enc_in_w   = (const float*)d_in[3];
  const float* enc_in_b   = (const float*)d_in[4];
  const float* enc_conv_w = (const float*)d_in[5];
  const float* enc_conv_b = (const float*)d_in[6];
  const float* enc_xproj_w= (const float*)d_in[7];
  const float* enc_dt_w   = (const float*)d_in[8];
  const float* enc_dt_b   = (const float*)d_in[9];
  const float* enc_A_log  = (const float*)d_in[10];
  const float* enc_D      = (const float*)d_in[11];
  const float* enc_out_w  = (const float*)d_in[12];
  const float* enc_out_b  = (const float*)d_in[13];
  const float* dec_in_w   = (const float*)d_in[14];
  const float* dec_in_b   = (const float*)d_in[15];
  const float* dec_conv_w = (const float*)d_in[16];
  const float* dec_conv_b = (const float*)d_in[17];
  const float* dec_xproj_w= (const float*)d_in[18];
  const float* dec_dt_w   = (const float*)d_in[19];
  const float* dec_dt_b   = (const float*)d_in[20];
  const float* dec_A_log  = (const float*)d_in[21];
  const float* dec_D      = (const float*)d_in[22];
  const float* dec_out_w  = (const float*)d_in[23];
  const float* dec_out_b  = (const float*)d_in[24];
  const float* enc_ann_w  = (const float*)d_in[25];
  const float* enc_ann_b  = (const float*)d_in[26];
  const float* dec_ann_w  = (const float*)d_in[27];
  const float* dec_ann_b  = (const float*)d_in[28];
  const float* fc_out_w   = (const float*)d_in[29];
  const float* fc_out_b   = (const float*)d_in[30];

  // ---- workspace bump allocator ----
  char* base = (char*)d_ws;
  size_t off = 0;
  auto alloc = [&](size_t bytes) -> void* {
    void* p = base + off;
    off = (off + bytes + 255) & ~(size_t)255;
    return p;
  };
  typedef unsigned short bf16_t;

  // bf16 weight buffers (padded where needed)
  bf16_t* w_fcin  = (bf16_t*)alloc((size_t)DMODEL * INSZ * 2);
  bf16_t* w_fcout = (bf16_t*)alloc((size_t)INSZ * DMODEL * 2);
  bf16_t* w_ann_e = (bf16_t*)alloc((size_t)16 * DMODEL * 2);       // (10,512)->(16,512)
  bf16_t* w_ann_d = (bf16_t*)alloc((size_t)DMODEL * 32 * 2);       // (512,10)->(512,32)
  bf16_t* w_in_e  = (bf16_t*)alloc((size_t)2 * DINNER * DMODEL * 2);
  bf16_t* w_xp_e  = (bf16_t*)alloc((size_t)64 * DINNER * 2);
  bf16_t* w_dt_e  = (bf16_t*)alloc((size_t)DINNER * DTRANK * 2);
  bf16_t* w_out_e = (bf16_t*)alloc((size_t)DMODEL * DINNER * 2);
  bf16_t* w_in_d  = (bf16_t*)alloc((size_t)2 * DINNER * DMODEL * 2);
  bf16_t* w_xp_d  = (bf16_t*)alloc((size_t)64 * DINNER * 2);
  bf16_t* w_dt_d  = (bf16_t*)alloc((size_t)DINNER * DTRANK * 2);
  bf16_t* w_out_d = (bf16_t*)alloc((size_t)DMODEL * DINNER * 2);

  // activation buffers (reused by both mamba blocks)
  bf16_t* xbf    = (bf16_t*)alloc((size_t)TOK * INSZ * 2);
  float*  act512 = (float*)alloc((size_t)TOK * DMODEL * 4);
  bf16_t* hbf    = (bf16_t*)alloc((size_t)TOK * DMODEL * 2);
  float*  xz     = (float*)alloc((size_t)TOK * 2 * DINNER * 4);
  float*  uc     = (float*)alloc((size_t)TOK * DINNER * 4);
  bf16_t* ucbf   = (bf16_t*)alloc((size_t)TOK * DINNER * 2);
  float*  xdbl   = (float*)alloc((size_t)TOK * 64 * 4);
  bf16_t* dtbf   = (bf16_t*)alloc((size_t)TOK * DTRANK * 2);
  float*  delta  = (float*)alloc((size_t)TOK * DINNER * 4);
  float*  ycore  = (float*)alloc((size_t)TOK * DINNER * 4);
  bf16_t* ygbf   = (bf16_t*)alloc((size_t)TOK * DINNER * 2);
  float*  bott   = (float*)alloc((size_t)TOK * 16 * 4);
  bf16_t* bottbf = (bf16_t*)alloc((size_t)TOK * 32 * 2);

  // ---- launch helpers ----
  auto cvt = [&](const float* src, bf16_t* dst, int rows, int srows,
                 int sstride, int soff, int cols, int dcols) {
    long long total = (long long)rows * dcols;
    int blocks = (int)((total + 255) / 256);
    hipLaunchKernelGGL(cvt_pad_bf16, dim3(blocks), dim3(256), 0, stream,
                       src, dst, rows, srows, sstride, soff, cols, dcols);
  };
  auto gemm = [&](const bf16_t* A, const bf16_t* W, const float* bias,
                  float* C, int M, int N, int K, int biasN, int epi) {
    dim3 grid(M / MB, (N / 16 + 31) / 32);
    hipLaunchKernelGGL(gemm_bf16_wmma, grid, dim3(256), 0, stream,
                       A, W, bias, C, M, N, K, biasN, epi);
  };

  // ---- convert weights to bf16 (with zero padding for bottleneck dims) ----
  cvt(fc_in_w,    w_fcin,  DMODEL, DMODEL, INSZ,   0, INSZ,   INSZ);
  cvt(fc_out_w,   w_fcout, INSZ,   INSZ,   DMODEL, 0, DMODEL, DMODEL);
  cvt(enc_ann_w,  w_ann_e, 16,     BOTTLE, DMODEL, 0, DMODEL, DMODEL);  // pad rows 10->16
  cvt(dec_ann_w,  w_ann_d, DMODEL, DMODEL, BOTTLE, 0, BOTTLE, 32);      // pad K 10->32
  cvt(enc_in_w,   w_in_e,  2*DINNER, 2*DINNER, DMODEL, 0, DMODEL, DMODEL);
  cvt(enc_xproj_w,w_xp_e,  64,     64,     DINNER, 0, DINNER, DINNER);
  cvt(enc_dt_w,   w_dt_e,  DINNER, DINNER, DTRANK, 0, DTRANK, DTRANK);
  cvt(enc_out_w,  w_out_e, DMODEL, DMODEL, DINNER, 0, DINNER, DINNER);
  cvt(dec_in_w,   w_in_d,  2*DINNER, 2*DINNER, DMODEL, 0, DMODEL, DMODEL);
  cvt(dec_xproj_w,w_xp_d,  64,     64,     DINNER, 0, DINNER, DINNER);
  cvt(dec_dt_w,   w_dt_d,  DINNER, DINNER, DTRANK, 0, DTRANK, DTRANK);
  cvt(dec_out_w,  w_out_d, DMODEL, DMODEL, DINNER, 0, DINNER, DINNER);

  const int elemBlocksDI = (TOK * DINNER) / 256;

  // one full mamba block: hbf (bf16 (TOK,DMODEL)) -> act512 (fp32 (TOK,DMODEL))
  auto mamba = [&](const bf16_t* win, const float* inb,
                   const float* convw, const float* convb,
                   const bf16_t* wxp,
                   const bf16_t* wdt, const float* dtb,
                   const float* Alog, const float* Dp,
                   const bf16_t* wout, const float* outb) {
    gemm(hbf, win, inb, xz, TOK, 2 * DINNER, DMODEL, 2 * DINNER, 0);
    hipLaunchKernelGGL(conv_silu_kernel, dim3(elemBlocksDI), dim3(256), 0, stream,
                       xz, convw, convb, uc);
    cvt(uc, ucbf, TOK, TOK, DINNER, 0, DINNER, DINNER);
    gemm(ucbf, wxp, nullptr, xdbl, TOK, 64, DINNER, 0, 0);
    cvt(xdbl, dtbf, TOK, TOK, 64, 0, DTRANK, DTRANK);
    gemm(dtbf, wdt, dtb, delta, TOK, DINNER, DTRANK, DINNER, 1);   // softplus
    hipLaunchKernelGGL(scan_kernel, dim3(BATCH_ * (DINNER / 256)), dim3(256), 0, stream,
                       xdbl, delta, uc, Alog, ycore);
    hipLaunchKernelGGL(gate_kernel, dim3(elemBlocksDI), dim3(256), 0, stream,
                       ycore, uc, xz, Dp, ygbf);
    gemm(ygbf, wout, outb, act512, TOK, DMODEL, DINNER, DMODEL, 0);
  };

  // ---- pipeline ----
  cvt(x, xbf, TOK, TOK, INSZ, 0, INSZ, INSZ);
  gemm(xbf, w_fcin, fc_in_b, act512, TOK, DMODEL, INSZ, DMODEL, 0);  // fc_in
  cvt(act512, hbf, TOK, TOK, DMODEL, 0, DMODEL, DMODEL);

  mamba(w_in_e, enc_in_b, enc_conv_w, enc_conv_b, w_xp_e,
        w_dt_e, enc_dt_b, enc_A_log, enc_D, w_out_e, enc_out_b);
  cvt(act512, hbf, TOK, TOK, DMODEL, 0, DMODEL, DMODEL);

  gemm(hbf, w_ann_e, enc_ann_b, bott, TOK, 16, DMODEL, BOTTLE, 0);   // enc_ann (pad N)
  cvt(bott, bottbf, TOK, TOK, 16, 0, 16, 32);                        // pad K 16->32
  gemm(bottbf, w_ann_d, dec_ann_b, act512, TOK, DMODEL, 32, DMODEL, 0); // dec_ann
  cvt(act512, hbf, TOK, TOK, DMODEL, 0, DMODEL, DMODEL);

  mamba(w_in_d, dec_in_b, dec_conv_w, dec_conv_b, w_xp_d,
        w_dt_d, dec_dt_b, dec_A_log, dec_D, w_out_d, dec_out_b);
  cvt(act512, hbf, TOK, TOK, DMODEL, 0, DMODEL, DMODEL);

  gemm(hbf, w_fcout, fc_out_b, (float*)d_out, TOK, INSZ, DMODEL, INSZ, 0); // fc_out
}
